// GNNNodeClassifier_88038239634290
// MI455X (gfx1250) — matrix operand order repaired and checked by reading, not compiled, not run on hardware
//
#include <hip/hip_runtime.h>

typedef __bf16 bf16_t;
typedef __bf16 v16bf __attribute__((ext_vector_type(16)));
typedef __bf16 v8bf  __attribute__((ext_vector_type(8)));
typedef float  v8f   __attribute__((ext_vector_type(8)));

#define H256 256

// Branch-free exact-GELU: 0.5*x*(1+erf(x/sqrt(2))) with A&S 7.1.26 erf
// (max abs err ~1.5e-7 — far below the bf16 GEMM noise floor). Avoids the
// divergent saveexec region libm erff generates.
__device__ __forceinline__ float gelu_exact(float x) {
    float z = x * 0.7071067811865475f;
    float a = fabsf(z);
    float t = __builtin_amdgcn_rcpf(fmaf(0.3275911f, a, 1.0f));
    float p = fmaf(1.061405429f, t, -1.453152027f);
    p = fmaf(p, t, 1.421413741f);
    p = fmaf(p, t, -0.284496736f);
    p = fmaf(p, t, 0.254829592f);
    p = p * t;
    float y = fmaf(-p, __expf(-a * a), 1.0f);   // erf(|z|)
    return 0.5f * x * (1.0f + copysignf(y, z));
}

// ---- A fragment: 16x32 bf16, row-major source X[row][k], stride K ----
// lane l (l<16): M=l,     K = kbase+{0..7}, kbase+16+{0..7}
// lane l (>=16): M=l-16,  K shifted by +8
template <int K>
__device__ __forceinline__ v16bf load_a_frag(const bf16_t* __restrict__ X,
                                             int row, int kbase, int lane) {
    int k0 = kbase + ((lane >> 4) << 3);
    const bf16_t* p = X + (size_t)row * K + k0;
    v8bf lo = *(const v8bf*)(p);
    v8bf hi = *(const v8bf*)(p + 16);
    v16bf a;
#pragma unroll
    for (int i = 0; i < 8; ++i) { a[i] = lo[i]; a[8 + i] = hi[i]; }
    return a;
}

// ---- A fragment from per-wave LDS tile h[rows][256] bf16 row-major ----
__device__ __forceinline__ v16bf load_a_frag_lds(const bf16_t* h, int rowbase,
                                                 int kbase, int lane) {
    int row = rowbase + (lane & 15);
    int k0 = kbase + ((lane >> 4) << 3);
    const bf16_t* p = h + row * H256 + k0;
    v8bf lo = *(const v8bf*)(p);
    v8bf hi = *(const v8bf*)(p + 16);
    v16bf a;
#pragma unroll
    for (int i = 0; i < 8; ++i) { a[i] = lo[i]; a[8 + i] = hi[i]; }
    return a;
}

// ---- B fragment: 32x16 bf16 from transposed weights Wt[n][k] (stride K) ----
// lanes 0-15: col = lane,    K = kbase+{0..7}, kbase+16+{0..7}
// lanes 16-31: col = lane-16, K shifted by +8
template <int K>
__device__ __forceinline__ v16bf load_b_frag(const bf16_t* __restrict__ Wt,
                                             int colbase, int kbase, int lane) {
    int n = colbase + (lane & 15);
    int k0 = kbase + ((lane >> 4) << 3);
    const bf16_t* p = Wt + (size_t)n * K + k0;
    v8bf lo = *(const v8bf*)(p);
    v8bf hi = *(const v8bf*)(p + 16);
    v16bf b;
#pragma unroll
    for (int i = 0; i < 8; ++i) { b[i] = lo[i]; b[8 + i] = hi[i]; }
    return b;
}

// =====================================================================
// Fused GEMM + bias + GELU:  Y[M,256] = gelu(X[M,K] @ W + b)
// X bf16 row-major, Wt = W^T bf16 [256][K].  outb (bf16) / outf (f32) optional.
// 8 waves/block; each wave owns RG row-groups of 16 rows, so every B
// fragment load feeds RG WMMAs (doubles arithmetic intensity at RG=2).
// =====================================================================
template <int K, int RG>
__global__ void __launch_bounds__(256) gemm_ffn(const bf16_t* __restrict__ X,
                                                const bf16_t* __restrict__ Wt,
                                                const float* __restrict__ bias,
                                                bf16_t* __restrict__ outb,
                                                float* __restrict__ outf, int M) {
    const int lane = threadIdx.x & 31;
    const int wave = threadIdx.x >> 5;
    const int rowbase = (blockIdx.x * 8 + wave) * (16 * RG);
    if (rowbase >= M) return;  // wave-uniform exit: EXEC all-ones for WMMA

    constexpr int KT = K / 32;
    v16bf afr[RG][KT];
#pragma unroll
    for (int g = 0; g < RG; ++g) {
        int r = rowbase + g * 16 + (lane & 15);
        int rload = (r < M) ? r : (M - 1);
#pragma unroll
        for (int kt = 0; kt < KT; ++kt)
            afr[g][kt] = load_a_frag<K>(X, rload, kt * 32, lane);
    }

    const int hi8 = (lane >> 4) << 3;
    for (int nt = 0; nt < 16; ++nt) {
        v8f acc[RG];
#pragma unroll
        for (int g = 0; g < RG; ++g) acc[g] = (v8f){};
#pragma unroll
        for (int kt = 0; kt < KT; ++kt) {
            v16bf b = load_b_frag<K>(Wt, nt * 16, kt * 32, lane);
#pragma unroll
            for (int g = 0; g < RG; ++g)
                acc[g] = __builtin_amdgcn_wmma_f32_16x16x32_bf16(
                    false, afr[g][kt], false, b, (short)0, acc[g], false, false);
        }
        int col = nt * 16 + (lane & 15);
        float bv = bias[col];
#pragma unroll
        for (int g = 0; g < RG; ++g) {
#pragma unroll
            for (int v = 0; v < 8; ++v) {
                int row = rowbase + g * 16 + v + hi8;  // D: lanes>=16 hold M=v+8
                if (row < M) {
                    float y = gelu_exact(acc[g][v] + bv);
                    if (outb) outb[(size_t)row * H256 + col] = (bf16_t)y;
                    if (outf) outf[(size_t)row * H256 + col] = y;
                }
            }
        }
    }
}

// =====================================================================
// Fused message kernel, 32-edge tiles (one wave each, 4 waves/block):
//   h  = gelu(x[nbr] @ W1 + b1)    (layer1 -> 16KB LDS tile, bf16)
//   m  = gelu(h @ W2 + b2) * w[e]  (layer2)
//   agg[tgt[e]] += m               (f32 global atomics, L2-resident)
// Avoids the ~800MB [E,256] intermediate; every B fragment feeds 2 WMMAs.
// =====================================================================
__global__ void __launch_bounds__(128) msg_kernel(const bf16_t* __restrict__ xb,
                                                  const bf16_t* __restrict__ W1t,
                                                  const float* __restrict__ b1,
                                                  const bf16_t* __restrict__ W2t,
                                                  const float* __restrict__ b2,
                                                  const int* __restrict__ tgt,
                                                  const int* __restrict__ nbr,
                                                  const float* __restrict__ ew,
                                                  const float* __restrict__ wsum,
                                                  float* __restrict__ agg, int E) {
    __shared__ bf16_t hst[4][32 * H256];  // 4 waves * 16KB = 64KB LDS
    const int lane = threadIdx.x & 31;
    const int wave = threadIdx.x >> 5;
    bf16_t* h = &hst[wave][0];
    const float inv = 1.0f / wsum[0];
    const int myrow = lane & 15;
    const int hi8 = (lane >> 4) << 3;

    const int ntiles = (E + 31) / 32;
    const int gw = blockIdx.x * 4 + wave;
    const int nw = gridDim.x * 4;

    for (int t = gw; t < ntiles; t += nw) {  // wave-uniform loop, no block syncs
        const int e0 = t * 32;

        // gfx1250 prefetch of the next tile's edge metadata (global_prefetch_b8)
        if (t + nw < ntiles) {
            int en = (t + nw) * 32;
            __builtin_prefetch(&nbr[en], 0, 1);
            __builtin_prefetch(&tgt[en], 0, 1);
            __builtin_prefetch(&ew[en], 0, 1);
        }

        int nb[2];
#pragma unroll
        for (int g = 0; g < 2; ++g) {
            int em = e0 + g * 16 + myrow;
            if (em >= E) em = E - 1;
            nb[g] = nbr[em];
        }

        // ---- layer 1: A fragments gathered from xb[nbr] ----
        v16bf afr[2][8];
#pragma unroll
        for (int g = 0; g < 2; ++g)
#pragma unroll
            for (int kt = 0; kt < 8; ++kt)
                afr[g][kt] = load_a_frag<H256>(xb, nb[g], kt * 32, lane);

        for (int nt = 0; nt < 16; ++nt) {
            v8f acc[2] = {(v8f){}, (v8f){}};
#pragma unroll
            for (int kt = 0; kt < 8; ++kt) {
                v16bf b = load_b_frag<H256>(W1t, nt * 16, kt * 32, lane);
#pragma unroll
                for (int g = 0; g < 2; ++g)
                    acc[g] = __builtin_amdgcn_wmma_f32_16x16x32_bf16(
                        false, afr[g][kt], false, b, (short)0, acc[g], false, false);
            }
            int col = nt * 16 + (lane & 15);
            float bv = b1[col];
#pragma unroll
            for (int g = 0; g < 2; ++g)
#pragma unroll
                for (int v = 0; v < 8; ++v) {
                    int row = g * 16 + v + hi8;
                    h[row * H256 + col] = (bf16_t)gelu_exact(acc[g][v] + bv);
                }
        }
        // per-wave LDS tile: DS ops are in-order within a wave; no barrier needed

        // ---- layer 2: A fragments from the LDS hidden tile (32 rows) ----
        v16bf a2[2][8];
#pragma unroll
        for (int g = 0; g < 2; ++g)
#pragma unroll
            for (int kt = 0; kt < 8; ++kt)
                a2[g][kt] = load_a_frag_lds(h, g * 16, kt * 32, lane);

        // hoist per-row edge metadata (rows g*16 + v + hi8)
        int t16[2][8];
        float w16[2][8];
#pragma unroll
        for (int g = 0; g < 2; ++g)
#pragma unroll
            for (int v = 0; v < 8; ++v) {
                int e = e0 + g * 16 + v + hi8;
                int ec = (e < E) ? e : (E - 1);
                t16[g][v] = tgt[ec];
                w16[g][v] = ew[ec] * inv;
            }

        for (int nt = 0; nt < 16; ++nt) {
            v8f acc[2] = {(v8f){}, (v8f){}};
#pragma unroll
            for (int kt = 0; kt < 8; ++kt) {
                v16bf b = load_b_frag<H256>(W2t, nt * 16, kt * 32, lane);
#pragma unroll
                for (int g = 0; g < 2; ++g)
                    acc[g] = __builtin_amdgcn_wmma_f32_16x16x32_bf16(
                        false, a2[g][kt], false, b, (short)0, acc[g], false, false);
            }
            int col = nt * 16 + (lane & 15);
            float bv = b2[col];
#pragma unroll
            for (int g = 0; g < 2; ++g)
#pragma unroll
                for (int v = 0; v < 8; ++v) {
                    int e = e0 + g * 16 + v + hi8;
                    if (e < E) {
                        float m = gelu_exact(acc[g][v] + bv) * w16[g][v];
                        atomicAdd(&agg[(size_t)t16[g][v] * H256 + col], m);
                    }
                }
        }
    }
}

// ---- edge-weight sum (block partials + one atomic per block) ----
__global__ void wsum_kernel(const float* __restrict__ ew, float* __restrict__ sum, int E) {
    __shared__ float red[8];
    float s = 0.f;
    for (int i = blockIdx.x * blockDim.x + threadIdx.x; i < E; i += gridDim.x * blockDim.x)
        s += ew[i];
    for (int off = 16; off; off >>= 1) s += __shfl_down(s, off, 32);
    if ((threadIdx.x & 31) == 0) red[threadIdx.x >> 5] = s;
    __syncthreads();
    if (threadIdx.x == 0) {
        float t = 0.f;
        for (int k = 0; k < 8; ++k) t += red[k];
        atomicAdd(sum, t);
    }
}

// ---- f32 W[K][N] row-major -> bf16 Wt[N][K] ----
__global__ void cvt_w_kernel(const float* __restrict__ W, bf16_t* __restrict__ Wt,
                             int K, int N) {
    int idx = blockIdx.x * blockDim.x + threadIdx.x;
    if (idx < K * N) {
        int k = idx / N, n = idx - k * N;
        Wt[(size_t)n * K + k] = (bf16_t)W[idx];
    }
}

// ---- f32 -> bf16 elementwise ----
__global__ void cvt_kernel(const float* __restrict__ src, bf16_t* __restrict__ dst, int n) {
    int i = blockIdx.x * blockDim.x + threadIdx.x;
    if (i < n) dst[i] = (bf16_t)src[i];
}

// ---- hb[i] = [ xb[i] : bf16(agg[i]) ]  (N x 512) ----
__global__ void concat_kernel(const bf16_t* __restrict__ xb, const float* __restrict__ agg,
                              bf16_t* __restrict__ hb, int N) {
    int idx = blockIdx.x * blockDim.x + threadIdx.x;
    if (idx < N * H256) {
        int i = idx / H256, j = idx - i * H256;
        hb[(size_t)i * 512 + j] = xb[idx];
        hb[(size_t)i * 512 + H256 + j] = (bf16_t)agg[idx];
    }
}

// ---- per-row l2-normalize(x1) + skip(xf) -> bf16 post-input ----
__global__ void __launch_bounds__(256) norm_skip_kernel(const float* __restrict__ x1,
                                                        const float* __restrict__ xf,
                                                        bf16_t* __restrict__ outb, int N) {
    __shared__ float red[8];
    __shared__ float tot;
    int i = blockIdx.x;
    int j = threadIdx.x;
    float v = x1[(size_t)i * H256 + j];
    float s = v * v;
    for (int off = 16; off; off >>= 1) s += __shfl_down(s, off, 32);
    if ((threadIdx.x & 31) == 0) red[threadIdx.x >> 5] = s;
    __syncthreads();
    if (threadIdx.x == 0) {
        float t = 0.f;
        for (int k = 0; k < 8; ++k) t += red[k];
        tot = t;
    }
    __syncthreads();
    float r = rsqrtf(fmaxf(tot, 1e-12f));
    outb[(size_t)i * H256 + j] = (bf16_t)(v * r + xf[(size_t)i * H256 + j]);
}

// ---- logits: out[B,4] = xfin[nidx] @ log_W + log_b  (tiny GEMV) ----
__global__ void logits_kernel(const float* __restrict__ xfin, const int* __restrict__ nidx,
                              const float* __restrict__ W, const float* __restrict__ b,
                              float* __restrict__ out, int B) {
    int t = blockIdx.x * blockDim.x + threadIdx.x;
    if (t >= B * 4) return;
    int i = t >> 2, c = t & 3;
    const float* row = xfin + (size_t)nidx[i] * H256;
    float s = b[c];
#pragma unroll 4
    for (int k = 0; k < H256; ++k) s += row[k] * W[k * 4 + c];
    out[t] = s;
}

extern "C" void kernel_launch(void* const* d_in, const int* in_sizes, int n_in,
                              void* d_out, int out_size, void* d_ws, size_t ws_size,
                              hipStream_t stream) {
    const float* node_features = (const float*)d_in[0];
    const int*   edges         = (const int*)d_in[1];
    const float* edge_weights  = (const float*)d_in[2];
    const int*   node_indices  = (const int*)d_in[3];
    const float* pre_W1  = (const float*)d_in[4];
    const float* pre_b1  = (const float*)d_in[5];
    const float* pre_W2  = (const float*)d_in[6];
    const float* pre_b2  = (const float*)d_in[7];
    const float* prep_W1 = (const float*)d_in[8];
    const float* prep_b1 = (const float*)d_in[9];
    const float* prep_W2 = (const float*)d_in[10];
    const float* prep_b2 = (const float*)d_in[11];
    const float* upd_W1  = (const float*)d_in[12];
    const float* upd_b1  = (const float*)d_in[13];
    const float* upd_W2  = (const float*)d_in[14];
    const float* upd_b2  = (const float*)d_in[15];
    const float* post_W1 = (const float*)d_in[16];
    const float* post_b1 = (const float*)d_in[17];
    const float* post_W2 = (const float*)d_in[18];
    const float* post_b2 = (const float*)d_in[19];
    const float* log_W   = (const float*)d_in[20];
    const float* log_b   = (const float*)d_in[21];

    const int N = in_sizes[0] / H256;
    const int E = in_sizes[2];
    const int B = in_sizes[3];
    const int* tgt = edges;
    const int* nbr = edges + E;

    // ---------- workspace carve-up (all 256B aligned) ----------
    char* ws = (char*)d_ws;
    size_t off = 0;
    auto take = [&](size_t bytes) -> char* {
        char* p = ws + off;
        off = (off + bytes + 255) & ~(size_t)255;
        return p;
    };
    float*  sumw    = (float*)take(4);
    bf16_t* preW1t  = (bf16_t*)take((size_t)256 * 256 * 2);
    bf16_t* preW2t  = (bf16_t*)take((size_t)256 * 256 * 2);
    bf16_t* prepW1t = (bf16_t*)take((size_t)256 * 256 * 2);
    bf16_t* prepW2t = (bf16_t*)take((size_t)256 * 256 * 2);
    bf16_t* updW1t  = (bf16_t*)take((size_t)512 * 256 * 2);
    bf16_t* updW2t  = (bf16_t*)take((size_t)256 * 256 * 2);
    bf16_t* postW1t = (bf16_t*)take((size_t)256 * 256 * 2);
    bf16_t* postW2t = (bf16_t*)take((size_t)256 * 256 * 2);
    bf16_t* featb   = (bf16_t*)take((size_t)N * H256 * 2);  // reused as post-input
    bf16_t* tmpb    = (bf16_t*)take((size_t)N * H256 * 2);  // FFN hidden
    bf16_t* xb      = (bf16_t*)take((size_t)N * H256 * 2);
    float*  xf      = (float*)take((size_t)N * H256 * 4);
    float*  agg     = (float*)take((size_t)N * H256 * 4);   // reused as x1 (f32)
    bf16_t* hb      = (bf16_t*)take((size_t)N * 512 * 2);   // reused as xfinal (f32)

    // ---------- zero accumulators (graph-capture legal) ----------
    hipMemsetAsync(sumw, 0, 4, stream);
    hipMemsetAsync(agg, 0, (size_t)N * H256 * 4, stream);

    // ---------- edge-weight sum ----------
    wsum_kernel<<<256, 256, 0, stream>>>(edge_weights, sumw, E);

    // ---------- weight transpose/convert to bf16 ----------
    {
        int g = (256 * 256 + 255) / 256;
        cvt_w_kernel<<<g, 256, 0, stream>>>(pre_W1, preW1t, 256, 256);
        cvt_w_kernel<<<g, 256, 0, stream>>>(pre_W2, preW2t, 256, 256);
        cvt_w_kernel<<<g, 256, 0, stream>>>(prep_W1, prepW1t, 256, 256);
        cvt_w_kernel<<<g, 256, 0, stream>>>(prep_W2, prepW2t, 256, 256);
        cvt_w_kernel<<<(512 * 256 + 255) / 256, 256, 0, stream>>>(upd_W1, updW1t, 512, 256);
        cvt_w_kernel<<<g, 256, 0, stream>>>(upd_W2, updW2t, 256, 256);
        cvt_w_kernel<<<g, 256, 0, stream>>>(post_W1, postW1t, 256, 256);
        cvt_w_kernel<<<g, 256, 0, stream>>>(post_W2, postW2t, 256, 256);
    }

    // ---------- node features -> bf16 ----------
    cvt_kernel<<<((N * H256) + 255) / 256, 256, 0, stream>>>(node_features, featb, N * H256);

    const int gemm_blocks_rg2 = (N + 255) / 256;  // 8 waves * 32 rows
    const int gemm_blocks_rg1 = (N + 127) / 128;  // 8 waves * 16 rows

    // ---------- preprocess FFN: x = ffn(features) ----------
    gemm_ffn<256, 2><<<gemm_blocks_rg2, 256, 0, stream>>>(featb, preW1t, pre_b1, tmpb, nullptr, N);
    gemm_ffn<256, 2><<<gemm_blocks_rg2, 256, 0, stream>>>(tmpb, preW2t, pre_b2, xb, xf, N);

    // ---------- fused message FFN + weighted scatter ----------
    {
        int ntiles = (E + 31) / 32;
        int blocks = (ntiles + 3) / 4;
        msg_kernel<<<blocks, 128, 0, stream>>>(xb, prepW1t, prep_b1, prepW2t, prep_b2,
                                               tgt, nbr, edge_weights, sumw, agg, E);
    }

    // ---------- concat [x : agg] ----------
    concat_kernel<<<((N * H256) + 255) / 256, 256, 0, stream>>>(xb, agg, hb, N);

    // ---------- update FFN (K=512 then K=256), x1 to f32 (reuse agg) ----------
    gemm_ffn<512, 1><<<gemm_blocks_rg1, 256, 0, stream>>>(hb, updW1t, upd_b1, tmpb, nullptr, N);
    gemm_ffn<256, 2><<<gemm_blocks_rg2, 256, 0, stream>>>(tmpb, updW2t, upd_b2, nullptr, agg, N);

    // ---------- l2-normalize + skip -> bf16 (reuse featb) ----------
    norm_skip_kernel<<<N, 256, 0, stream>>>(agg, xf, featb, N);

    // ---------- postprocess FFN, final x in f32 (reuse hb) ----------
    gemm_ffn<256, 2><<<gemm_blocks_rg2, 256, 0, stream>>>(featb, postW1t, post_b1, tmpb, nullptr, N);
    gemm_ffn<256, 2><<<gemm_blocks_rg2, 256, 0, stream>>>(tmpb, postW2t, post_b2, nullptr, (float*)hb, N);

    // ---------- logits ----------
    logits_kernel<<<(B * 4 + 255) / 256, 256, 0, stream>>>((const float*)hb, node_indices,
                                                           log_W, log_b, (float*)d_out, B);
}
